// RecSys_64415919505726
// MI455X (gfx1250) — matrix operand order repaired and checked by reading, not compiled, not run on hardware
//
#include <hip/hip_runtime.h>

typedef __attribute__((ext_vector_type(16))) __bf16 v16bf;
typedef __attribute__((ext_vector_type(8)))  float  v8f;
typedef __attribute__((ext_vector_type(4)))  float  f4;

#define NTOT   128         // every GEMM here has N = E = 128
#define WAVES  8
#define BLOCK  (WAVES * 32)
#define MSLAB  32          // rows per workgroup (2 x 16-row WMMA tiles per wave)
#define MAXSPLIT 32        // P partial buffer sized for this

// Unaligned-safe 16-byte load; gfx1250 supports unaligned b128 accesses and
// the backend merges this into a single global_load_b128.
__device__ __forceinline__ f4 ld4(const float* p)
{
    f4 v;
    __builtin_memcpy(&v, p, 16);
    return v;
}

// Pack four float4 chunks into one 16-element bf16 fragment (lowered to
// v_cvt_pk_bf16_f32 pairs).
__device__ __forceinline__ v16bf cvt16(f4 a, f4 b, f4 c, f4 d)
{
    v16bf r;
    #pragma unroll
    for (int j = 0; j < 4; ++j) {
        r[j]      = (__bf16)a[j];
        r[4 + j]  = (__bf16)b[j];
        r[8 + j]  = (__bf16)c[j];
        r[12 + j] = (__bf16)d[j];
    }
    return r;
}

// ---- guarded (zero-padded) fragment loaders for the K tail ----------------
// A 16x32 bf16 (ISA 7.12.2): lanes 0-15 hold K=[k0..k0+7],[k0+16..k0+23];
// lanes 16-31 hold K=[k0+8..k0+15],[k0+24..k0+31].
__device__ __forceinline__ v16bf load_a_guard(const float* __restrict__ base,
                                              long k0, int half, long K)
{
    v16bf a;
    #pragma unroll
    for (int j = 0; j < 8; ++j) {
        long ka = k0 + half * 8 + j;
        a[j]     = (__bf16)((ka      < K) ? base[ka]      : 0.0f);
        a[j + 8] = (__bf16)((ka + 16 < K) ? base[ka + 16] : 0.0f);
    }
    return a;
}
// B 32x16 bf16: lane = output column; lanes 0-15 hold K=[k0..k0+7],[k0+16..k0+23];
// lanes 16-31 hold K=[k0+8..k0+15],[k0+24..k0+31] (same as A).
__device__ __forceinline__ v16bf load_b_guard(const float* __restrict__ base,
                                              long k0, int half, long K)
{
    v16bf b;
    #pragma unroll
    for (int j = 0; j < 8; ++j) {
        long kb = k0 + half * 8 + j;
        b[j]     = (__bf16)((kb      < K) ? base[kb]      : 0.0f);
        b[j + 8] = (__bf16)((kb + 16 < K) ? base[kb + 16] : 0.0f);
    }
    return b;
}

__device__ __forceinline__ v8f wmma_bf16(v16bf a, v16bf b, v8f c)
{
    return __builtin_amdgcn_wmma_f32_16x16x32_bf16(
        /*neg_a=*/false, a, /*neg_b=*/false, b,
        /*c_mod=*/(short)0, c, /*reuse_a=*/false, /*reuse_b=*/false);
}

// C[M,128] (split-K partials) = A[M,K] * W[128,K]^T
// A row-major [M][K], W row-major [128][K].  Partials: P[split][M][128].
// Each wave: 2 M-tiles sharing one B fragment; block: 32(M) x 128(N) slab.
__global__ __launch_bounds__(BLOCK)
void gemm_bf16_wmma(const float* __restrict__ A,
                    const float* __restrict__ W,
                    float* __restrict__ P,
                    int M, int K, int steps_per_split)
{
    const int lane  = threadIdx.x & 31;
    const int wid   = threadIdx.x >> 5;
    const int m0    = blockIdx.x * MSLAB;
    const int split = blockIdx.y;
    const int n0    = wid * 16;

    const int half = lane >> 4;
    const int l15  = lane & 15;

    long kstart = (long)split * steps_per_split * 32;
    long kend   = kstart + (long)steps_per_split * 32;
    if (kend > (long)K) kend = (long)K;
    long kfast = kend & ~31L;            // kstart is a multiple of 32
    if (kfast < kstart) kfast = kstart;  // defensive (empty split)

    const int bcol = n0 + l15;           // output column == W row
    const float* __restrict__ A0 = A + (size_t)(m0 + l15) * K;
    const float* __restrict__ A1 = A0 + (size_t)16 * K;
    const float* __restrict__ Wb = W + (size_t)bcol * K;

    // per-lane base pointers including the half-wave K-phase
    const float* __restrict__ ap0 = A0 + half * 8;
    const float* __restrict__ ap1 = A1 + half * 8;
    const float* __restrict__ bp  = Wb + half * 8;

    v8f c0 = {}, c1 = {};

    // branch-free steady-state loop: phase 1 = 12 independent b128 loads,
    // phase 2 = bf16 packing, phase 3 = two WMMAs sharing the B fragment.
    for (long k0 = kstart; k0 < kfast; k0 += 32) {
        const float* b_ = bp  + k0;
        const float* x_ = ap0 + k0;
        const float* y_ = ap1 + k0;

        f4 rb0 = ld4(b_ + 0),  rb1 = ld4(b_ + 4);
        f4 rb2 = ld4(b_ + 16), rb3 = ld4(b_ + 20);
        f4 rx0 = ld4(x_ + 0),  rx1 = ld4(x_ + 4);
        f4 rx2 = ld4(x_ + 16), rx3 = ld4(x_ + 20);
        f4 ry0 = ld4(y_ + 0),  ry1 = ld4(y_ + 4);
        f4 ry2 = ld4(y_ + 16), ry3 = ld4(y_ + 20);

        v16bf b  = cvt16(rb0, rb1, rb2, rb3);
        v16bf a0 = cvt16(rx0, rx1, rx2, rx3);
        v16bf a1 = cvt16(ry0, ry1, ry2, ry3);

        c0 = wmma_bf16(a0, b, c0);
        c1 = wmma_bf16(a1, b, c1);
    }

    // peeled zero-padded tail (runs only when K is not a multiple of 32)
    if (kfast < kend) {
        v16bf b  = load_b_guard(Wb, kfast, half, (long)K);
        v16bf a0 = load_a_guard(A0, kfast, half, (long)K);
        v16bf a1 = load_a_guard(A1, kfast, half, (long)K);
        c0 = wmma_bf16(a0, b, c0);
        c1 = wmma_bf16(a1, b, c1);
    }

    // C/D layout: VGPR r -> M = r + 8*half, N = bcol   (ISA 7.12.2)
    float* __restrict__ Pout = P + ((size_t)split * M + m0) * NTOT;
    #pragma unroll
    for (int r = 0; r < 8; ++r) {
        int row = r + half * 8;
        Pout[(size_t)row        * NTOT + bcol] = c0[r];
        Pout[(size_t)(row + 16) * NTOT + bcol] = c1[r];
    }
}

// Deterministic fixed-order split-K reduction + bias + optional ReLU.
__global__ __launch_bounds__(256)
void reduce_bias(const float* __restrict__ P, int splitk, int M,
                 const float* __restrict__ bias,
                 float* __restrict__ dst, int ldc, int col_off, int relu)
{
    int idx = blockIdx.x * blockDim.x + threadIdx.x;
    if (idx >= M * NTOT) return;
    int b = idx / NTOT, n = idx % NTOT;
    float acc = bias[n];
    for (int s = 0; s < splitk; ++s)
        acc += P[((size_t)s * M + b) * NTOT + n];
    if (relu) acc = fmaxf(acc, 0.0f);
    dst[(size_t)b * ldc + col_off + n] = acc;
}

// In-place training-mode BatchNorm1d over the batch dim (biased variance).
__global__ __launch_bounds__(256)
void bn_cols(float* __restrict__ X, int M, int C,
             const float* __restrict__ gamma, const float* __restrict__ beta)
{
    __shared__ float s1[256];
    __shared__ float s2[256];
    const int col = blockIdx.x;
    const int t   = threadIdx.x;
    float sum = 0.0f, sq = 0.0f;
    for (int r = t; r < M; r += 256) {
        float v = X[(size_t)r * C + col];
        sum += v; sq += v * v;
    }
    s1[t] = sum; s2[t] = sq;
    __syncthreads();
    for (int o = 128; o > 0; o >>= 1) {
        if (t < o) { s1[t] += s1[t + o]; s2[t] += s2[t + o]; }
        __syncthreads();
    }
    const float mean  = s1[0] / (float)M;
    const float var   = s2[0] / (float)M - mean * mean;
    const float scale = rsqrtf(var + 1e-5f) * gamma[col];
    const float shift = beta[col] - mean * scale;
    for (int r = t; r < M; r += 256) {
        size_t i = (size_t)r * C + col;
        X[i] = X[i] * scale + shift;
    }
}

// out[b] = dot(U[b,:], V[b,:]), E = 128, one block per row.
__global__ __launch_bounds__(128)
void dot_rows(const float* __restrict__ U, const float* __restrict__ V,
              float* __restrict__ out, int E)
{
    __shared__ float s[128];
    const int b = blockIdx.x, t = threadIdx.x;
    s[t] = U[(size_t)b * E + t] * V[(size_t)b * E + t];
    __syncthreads();
    for (int o = 64; o > 0; o >>= 1) {
        if (t < o) s[t] += s[t + o];
        __syncthreads();
    }
    if (t == 0) out[b] = s[0];
}

static void run_gemm(const float* A, const float* W, const float* bias,
                     float* P, float* dst, int ldc, int col_off,
                     int M, int K, int splitk, int relu, hipStream_t stream)
{
    int steps = (K + 31) / 32;
    int sps   = (steps + splitk - 1) / splitk;
    dim3 grid(M / MSLAB, splitk);
    gemm_bf16_wmma<<<grid, BLOCK, 0, stream>>>(A, W, P, M, K, sps);
    int total = M * NTOT;
    reduce_bias<<<(total + 255) / 256, 256, 0, stream>>>(
        P, splitk, M, bias, dst, ldc, col_off, relu);
}

extern "C" void kernel_launch(void* const* d_in, const int* in_sizes, int n_in,
                              void* d_out, int out_size, void* d_ws, size_t ws_size,
                              hipStream_t stream)
{
    constexpr int B  = 512;
    constexpr int E  = 128;
    constexpr int KW = 372226, AU = 19065, MG = 28028, TG = 86037;

    const float* keywords  = (const float*)d_in[0];
    const float* following = (const float*)d_in[1];
    const float* author    = (const float*)d_in[2];
    const float* magazine  = (const float*)d_in[3];
    const float* tag       = (const float*)d_in[4];
    const float* W_kw = (const float*)d_in[5];   const float* b_kw = (const float*)d_in[6];
    const float* W_au = (const float*)d_in[7];   const float* b_au = (const float*)d_in[8];
    const float* W_mg = (const float*)d_in[9];   const float* b_mg = (const float*)d_in[10];
    const float* W_tg = (const float*)d_in[11];  const float* b_tg = (const float*)d_in[12];
    const float* g_u  = (const float*)d_in[13];  const float* be_u = (const float*)d_in[14];
    const float* W_u1 = (const float*)d_in[15];  const float* b_u1 = (const float*)d_in[16];
    const float* W_u2 = (const float*)d_in[17];  const float* b_u2 = (const float*)d_in[18];
    const float* g_i  = (const float*)d_in[19];  const float* be_i = (const float*)d_in[20];
    const float* W_i1 = (const float*)d_in[21];  const float* b_i1 = (const float*)d_in[22];
    const float* W_i2 = (const float*)d_in[23];  const float* b_i2 = (const float*)d_in[24];

    // Workspace carve-up (floats). P sized for MAXSPLIT=32 -> 8 MB.
    float* ws     = (float*)d_ws;
    float* P      = ws;                                   // [32][512][128]
    float* ue_pre = P      + (size_t)MAXSPLIT * B * NTOT; // [512, 256]
    float* ie_pre = ue_pre + (size_t)B * 2 * E;           // [512, 384]
    float* h      = ie_pre + (size_t)B * 3 * E;           // [512, 128] (reused)
    float* uef    = h      + (size_t)B * E;               // [512, 128]
    float* ief    = uef    + (size_t)B * E;               // [512, 128]
    (void)in_sizes; (void)n_in; (void)out_size; (void)ws_size;

    // --- embedding projections (split-K bf16 WMMA GEMMs) ---
    run_gemm(keywords,  W_kw, b_kw, P, ue_pre, 2 * E, 0,     B, KW, 32, 0, stream);
    run_gemm(following, W_au, b_au, P, ue_pre, 2 * E, E,     B, AU,  8, 0, stream);
    run_gemm(author,    W_au, b_au, P, ie_pre, 3 * E, 0,     B, AU,  8, 0, stream);
    run_gemm(magazine,  W_mg, b_mg, P, ie_pre, 3 * E, E,     B, MG,  8, 0, stream);
    run_gemm(tag,       W_tg, b_tg, P, ie_pre, 3 * E, 2 * E, B, TG, 16, 0, stream);

    // --- batchnorm (training mode, batch statistics) ---
    bn_cols<<<2 * E, 256, 0, stream>>>(ue_pre, B, 2 * E, g_u, be_u);
    bn_cols<<<3 * E, 256, 0, stream>>>(ie_pre, B, 3 * E, g_i, be_i);

    // --- user tower ---
    run_gemm(ue_pre, W_u1, b_u1, P, h,   E, 0, B, 2 * E, 1, 1, stream);  // ReLU
    run_gemm(h,      W_u2, b_u2, P, uef, E, 0, B, E,     1, 0, stream);

    // --- item tower ---
    run_gemm(ie_pre, W_i1, b_i1, P, h,   E, 0, B, 3 * E, 1, 1, stream);  // ReLU
    run_gemm(h,      W_i2, b_i2, P, ief, E, 0, B, E,     1, 0, stream);

    // --- final per-row dot product ---
    dot_rows<<<B, 128, 0, stream>>>(uef, ief, (float*)d_out, E);
}
